// DynamicGRU_661424964275
// MI455X (gfx1250) — matrix operand order repaired and checked by reading, not compiled, not run on hardware
//
#include <hip/hip_runtime.h>

#define Bsz 2048
#define Tt  128
#define Dd  256
#define Hh  256

typedef unsigned short u16;
typedef __attribute__((ext_vector_type(16))) __bf16 v16bf;
typedef __attribute__((ext_vector_type(8)))  float  v8f;

union Frag {
    v16bf bf;
    uint4 q[2];
};

__device__ __forceinline__ u16 f2bf(float f) {
    unsigned u = __builtin_bit_cast(unsigned, f);
    u += 0x7FFFu + ((u >> 16) & 1u);   // round-to-nearest-even
    return (u16)(u >> 16);
}

__device__ __forceinline__ unsigned pack2(float a, float b) {
    return (unsigned)f2bf(a) | ((unsigned)f2bf(b) << 16);
}

__device__ __forceinline__ float sigf(float x) {
    return 1.0f / (1.0f + __expf(-x));
}

__device__ __forceinline__ float tanh_fast(float x) {
    // tanh(x) = (1 - e^-2x) / (1 + e^-2x); one v_exp_f32 + one v_rcp path
    float e = __expf(-2.0f * x);
    return (1.0f - e) / (1.0f + e);
}

// One-time fp32 -> bf16 weight conversion into workspace (L2-resident thereafter).
__global__ void w_to_bf16(const float* __restrict__ Wx, const float* __restrict__ Wh,
                          u16* __restrict__ Wxb, u16* __restrict__ Whb, int n) {
    int i = blockIdx.x * blockDim.x + threadIdx.x;
    if (i < n) {
        Wxb[i] = f2bf(Wx[i]);
        Whb[i] = f2bf(Wh[i]);
    }
}

// Persistent AUGRU: one workgroup = 16 batch rows for all T steps.
// 16 waves; wave w owns hidden cols [16w,16w+16) of gates u,r,n.
//
// NOTE: Wxb/Whb/outSeq are deliberately NOT __restrict__: the outSeq stores in
// the gate phase must be assumed to alias the weights, which forbids the
// compiler from hoisting the (t-invariant) B-fragment loads out of the time
// loop. Hoisting them needs 384 VGPRs and caused massive scratch spilling in
// the previous revision; streaming them from L2 each step is the intended
// data path on MI455X (weights are 768 KB, fully L2-resident).
__global__ __launch_bounds__(512, 1)
void augru_wmma(const float* __restrict__ x, const float* __restrict__ attn,
                const int* __restrict__ lengths,
                const float* __restrict__ bx, const float* __restrict__ bh,
                const u16* Wxb, const u16* Whb,
                float* outSeq, float* __restrict__ outH) {
    __shared__ alignas(16) u16 xA[16 * Dd];   // x tile, bf16, row-major
    __shared__ alignas(16) u16 hA[16 * Hh];   // h tile, bf16, row-major

    const int tid     = threadIdx.x;
    const int wave    = tid >> 5;
    const int lane    = tid & 31;
    const int nIdx    = lane & 15;            // N within tile; also M for A frags
    const bool lowH   = (lane < 16);
    const int mAdd    = lowH ? 0 : 8;         // C/D: VGPR v holds M=v (lanes0-15) / v+8
    const int kloA    = lowH ? 0 : 8;         // A frag: lanes16-31 start at K+8
    const int khiB    = lowH ? 0 : 16;        // unused: B frags use kloA (same K order as A)
    const int col0    = wave * 16;            // hidden-column base for this wave
    const int jcol    = col0 + nIdx;          // this lane's hidden unit
    const int b0      = blockIdx.x * 16;

    // zero the hidden bf16 buffer (h0 = 0)
    {
        uint4 z; z.x = z.y = z.z = z.w = 0u;
        *(uint4*)&hA[tid * 8] = z;            // 512 threads * 8 u16 = 4096
    }

    float bxr[3], bhr[3];
#pragma unroll
    for (int g = 0; g < 3; ++g) {
        bxr[g] = bx[g * Hh + jcol];
        bhr[g] = bh[g * Hh + jcol];
    }
    int   lenm[8];
    float hreg[8];                            // f32 hidden state, fixed (m,j) per lane
#pragma unroll
    for (int v = 0; v < 8; ++v) {
        lenm[v] = lengths[b0 + v + mAdd];
        hreg[v] = 0.0f;
    }

    const int srow = tid >> 5;                // staging: 16 rows
    const int scol = (tid & 31) * 8;          // 32 threads/row * 8 floats

    const v8f zero8 = {0.f, 0.f, 0.f, 0.f, 0.f, 0.f, 0.f, 0.f};

    for (int t = 0; t < Tt; ++t) {
        // ---- stage x[b0..b0+15, t, :] into LDS as bf16 ----
        {
            const float* xp = x + ((size_t)(b0 + srow) * Tt + t) * Dd + scol;
            float4 f0 = *(const float4*)xp;
            float4 f1 = *(const float4*)(xp + 4);
            uint4 pk;
            pk.x = pack2(f0.x, f0.y);
            pk.y = pack2(f0.z, f0.w);
            pk.z = pack2(f1.x, f1.y);
            pk.w = pack2(f1.z, f1.w);
            *(uint4*)&xA[srow * Dd + scol] = pk;
            if (t + 1 < Tt) {
                // global_prefetch_b8 for next step's x rows (hide HBM latency)
                __builtin_prefetch(xp + Dd, 0, 3);
            }
        }
        __syncthreads();   // x staged; previous step's hA writes visible

        v8f ax[3] = {zero8, zero8, zero8};    // x·Wxᵀ partials for gates u,r,n
        v8f ah[3] = {zero8, zero8, zero8};    // h·Whᵀ partials (kept separate: n-gate)

        // ---- GEMM 1: x tile (16x256) x Wx^T columns ----
#pragma unroll
        for (int kk = 0; kk < Dd / 32; ++kk) {
            Frag a;
            const u16* ap = &xA[nIdx * Dd + kk * 32 + kloA];
            a.q[0] = *(const uint4*)ap;        // K +0..7   (rel. to lane half)
            a.q[1] = *(const uint4*)(ap + 16); // K +16..23
#pragma unroll
            for (int g = 0; g < 3; ++g) {
                Frag b;
                const u16* bp = Wxb + (size_t)(g * Hh + jcol) * Dd + kk * 32 + kloA;
                b.q[0] = *(const uint4*)bp;
                b.q[1] = *(const uint4*)(bp + 16);
                ax[g] = __builtin_amdgcn_wmma_f32_16x16x32_bf16(
                            false, a.bf, false, b.bf, (short)0, ax[g], false, false);
            }
        }
        // ---- GEMM 2: h tile (16x256) x Wh^T columns ----
#pragma unroll
        for (int kk = 0; kk < Hh / 32; ++kk) {
            Frag a;
            const u16* ap = &hA[nIdx * Hh + kk * 32 + kloA];
            a.q[0] = *(const uint4*)ap;
            a.q[1] = *(const uint4*)(ap + 16);
#pragma unroll
            for (int g = 0; g < 3; ++g) {
                Frag b;
                const u16* bp = Whb + (size_t)(g * Hh + jcol) * Hh + kk * 32 + kloA;
                b.q[0] = *(const uint4*)bp;
                b.q[1] = *(const uint4*)(bp + 16);
                ah[g] = __builtin_amdgcn_wmma_f32_16x16x32_bf16(
                            false, a.bf, false, b.bf, (short)0, ah[g], false, false);
            }
        }
        __syncthreads();   // all hA reads done before gate phase rewrites it

        // ---- gates entirely in registers (u,r,n share the lane/VGPR slot) ----
        float a_m[8];
#pragma unroll
        for (int v = 0; v < 8; ++v)
            a_m[v] = attn[(size_t)(b0 + v + mAdd) * Tt + t];

#pragma unroll
        for (int v = 0; v < 8; ++v) {
            const int m = v + mAdd;
            float u  = sigf(ax[0][v] + bxr[0] + ah[0][v] + bhr[0]) * a_m[v];
            float r  = sigf(ax[1][v] + bxr[1] + ah[1][v] + bhr[1]);
            float nn = tanh_fast(ax[2][v] + bxr[2] + r * (ah[2][v] + bhr[2]));
            float ho = hreg[v];
            float hy = ho + u * (nn - ho);
            bool valid = (t < lenm[v]);
            float hn = valid ? hy : ho;
            hreg[v] = hn;
            hA[m * Hh + jcol] = f2bf(hn);     // feed next step's A fragments
            outSeq[((size_t)(b0 + m) * Tt + t) * Hh + jcol] = valid ? hy : 0.0f;
        }
        // next iteration's first __syncthreads orders these hA writes vs. reads
    }

    // ---- final hidden state ----
#pragma unroll
    for (int v = 0; v < 8; ++v) {
        const int m = v + mAdd;
        outH[(size_t)(b0 + m) * Hh + jcol] = hreg[v];
    }
}

extern "C" void kernel_launch(void* const* d_in, const int* in_sizes, int n_in,
                              void* d_out, int out_size, void* d_ws, size_t ws_size,
                              hipStream_t stream) {
    (void)in_sizes; (void)n_in; (void)out_size; (void)ws_size;

    const float* x       = (const float*)d_in[0];   // [B,T,D]
    const float* attn    = (const float*)d_in[1];   // [B,T]
    const int*   lengths = (const int*)d_in[2];     // [B]
    const float* Wx      = (const float*)d_in[3];   // [3H,D]
    const float* bx      = (const float*)d_in[4];   // [3H]
    const float* Wh      = (const float*)d_in[5];   // [3H,H]
    const float* bh      = (const float*)d_in[6];   // [3H]

    float* outSeq = (float*)d_out;                          // [B,T,H]
    float* outH   = outSeq + (size_t)Bsz * Tt * Hh;         // [B,H]

    u16* Wxb = (u16*)d_ws;                 // 3H*D bf16
    u16* Whb = Wxb + 3 * Hh * Dd;          // 3H*H bf16

    const int nW = 3 * Hh * Dd;            // 196608 (== 3H*H as well)
    w_to_bf16<<<(nW + 255) / 256, 256, 0, stream>>>(Wx, Wh, Wxb, Whb, nW);

    augru_wmma<<<Bsz / 16, 512, 0, stream>>>(x, attn, lengths, bx, bh,
                                             Wxb, Whb, outSeq, outH);
}
